// GCN_49692771615418
// MI455X (gfx1250) — matrix operand order repaired and checked by reading, not compiled, not run on hardware
//
#include <hip/hip_runtime.h>
#include <hip/hip_bf16.h>
#include <math.h>

#define NN 50000
#define NE 800000
#define D  128
#define NC 10
#define NG 64
#define EPW 4   // edges per wave in the scatter kernel

typedef __attribute__((ext_vector_type(16))) _Float16 v16h;
typedef __attribute__((ext_vector_type(8)))  float    v8f;

// ---------------------------------------------------------------------------
// Degree / normalization
// ---------------------------------------------------------------------------
__global__ void k_deg_init(float* deg) {
    int i = blockIdx.x * blockDim.x + threadIdx.x;
    if (i < NN) deg[i] = 1.0f;                      // self-loop contributes 1
}

__global__ void k_deg_count(const int* __restrict__ col, float* __restrict__ deg) {
    int e = blockIdx.x * blockDim.x + threadIdx.x;
    if (e < NE) atomicAdd(&deg[col[e]], 1.0f);
}

__global__ void k_dinv(float* deg) {                // in place: deg -> rsqrt(deg)
    int i = blockIdx.x * blockDim.x + threadIdx.x;
    if (i < NN) deg[i] = rsqrtf(deg[i]);            // deg >= 1 always
}

// ---------------------------------------------------------------------------
// WMMA GEMM:  Hs[m,n] = (A[m,:] @ W[:,n]) * dinv[m]
//             Agg[m,n] = Hs[m,n] * dinv[m]        (self-loop init term)
// A: [M x 128] f32 row-major, W: [128 x 128] f32 row-major.
// Grid: ceil(M/32) blocks x 128 threads (4 waves). Each wave owns N-tiles
// {32w, 32w+16} and both M-tiles {m0, m0+16}: 64 WMMAs per block.
//
// W is staged in LDS pre-packed in B-fragment order so a B fragment is ONE
// contiguous 32-byte LDS read (2x ds_load_b128).
//
// Safe with Hs == A (per-row in-place): a block only reads its own 32 rows,
// and a barrier separates the K-loop reads from the epilogue stores.
// ---------------------------------------------------------------------------
__global__ __launch_bounds__(128)
void k_gemm_gcn(const float* __restrict__ A, const float* __restrict__ W,
                const float* __restrict__ dinv,
                float* __restrict__ Hs, float* __restrict__ Agg, int M) {
    __shared__ v16h Wp[8 * 4 * 32];                 // 32 KB packed fragments
    _Float16* Wph = (_Float16*)Wp;

    const int tid = threadIdx.x;
    // stage + pack: element (k,n) -> Wp[(tile*4+kk)*32 + khalf*16+ncol][e]
    #pragma unroll 4
    for (int i = tid; i < D * D; i += 128) {
        int k = i >> 7, n = i & 127;
        int tile = n >> 4, ncol = n & 15;
        int kk = k >> 5, kr = k & 31;
        int kh = kr >> 4, e = kr & 15;
        Wph[(((tile << 2) + kk) * 32 + (kh << 4) + ncol) * 16 + e] = (_Float16)W[i];
    }
    __syncthreads();

    const int wave  = tid >> 5;
    const int lane  = tid & 31;
    const int mrow  = lane & 15;
    const int khalf = lane >> 4;
    const int ncol  = lane & 15;
    const int m0    = blockIdx.x * 32;
    const int n0a   = wave * 32;
    const int n0b   = n0a + 16;
    const int ta    = wave * 2;                     // packed N-tile indices
    const int tb    = ta + 1;

    const int mA0 = m0 + mrow;                      // tile 0 row (always < M)
    int mA1 = m0 + 16 + mrow;                       // tile 1 row (may run off end)
    if (mA1 >= M) mA1 = M - 1;                      // clamp load, stores guarded
    const float* __restrict__ arow0 = A + (size_t)mA0 * D;
    const float* __restrict__ arow1 = A + (size_t)mA1 * D;

    v8f c00 = {}, c01 = {}, c10 = {}, c11 = {};
    #pragma unroll
    for (int kc = 0; kc < 4; ++kc) {
        const int kk = kc * 32;
        // --- A fragments: lane covers K run [kb,kb+16)
        const int kb = kk + khalf * 16;
        v16h a0, a1;
        #pragma unroll
        for (int j = 0; j < 8; ++j) {
            a0[j]     = (_Float16)arow0[kb + j];
            a0[j + 8] = (_Float16)arow0[kb + 8 + j];
            a1[j]     = (_Float16)arow1[kb + j];
            a1[j + 8] = (_Float16)arow1[kb + 8 + j];
        }
        // --- B fragments: single contiguous 32B LDS read per fragment
        v16h b0 = Wp[((ta << 2) + kc) * 32 + lane];
        v16h b1 = Wp[((tb << 2) + kc) * 32 + lane];

        c00 = __builtin_amdgcn_wmma_f32_16x16x32_f16(false, a0, false, b0,
                                                     (short)0, c00, false, false);
        c01 = __builtin_amdgcn_wmma_f32_16x16x32_f16(false, a0, false, b1,
                                                     (short)0, c01, false, false);
        c10 = __builtin_amdgcn_wmma_f32_16x16x32_f16(false, a1, false, b0,
                                                     (short)0, c10, false, false);
        c11 = __builtin_amdgcn_wmma_f32_16x16x32_f16(false, a1, false, b1,
                                                     (short)0, c11, false, false);
    }

    __syncthreads();   // all A reads done before in-place epilogue stores

    // D layout: VGPR r, lane -> m = r + 8*khalf, n = ncol
    #pragma unroll
    for (int r = 0; r < 8; ++r) {
        int m = m0 + r + 8 * khalf;
        {   // tile 0 rows: m < M guaranteed (M multiple of 16)
            float dv = dinv[m];
            size_t o0 = (size_t)m * D + n0a + ncol;
            size_t o1 = (size_t)m * D + n0b + ncol;
            float h0 = c00[r] * dv, h1 = c01[r] * dv;
            Hs[o0]  = h0;        Hs[o1]  = h1;
            Agg[o0] = h0 * dv;   Agg[o1] = h1 * dv;
        }
        int m2 = m + 16;
        if (m2 < M) {
            float dv = dinv[m2];
            size_t o0 = (size_t)m2 * D + n0a + ncol;
            size_t o1 = (size_t)m2 * D + n0b + ncol;
            float h0 = c10[r] * dv, h1 = c11[r] * dv;
            Hs[o0]  = h0;        Hs[o1]  = h1;
            Agg[o0] = h0 * dv;   Agg[o1] = h1 * dv;
        }
    }
}

// ---------------------------------------------------------------------------
// Edge scatter: Agg[col] += Hs[row] * dinv[col]   (Hs pre-scaled by dinv[row])
// One wave per EPW edges, 4 floats per lane. Edge indices are wave-uniform ->
// scalar loads. The next edge's 512B source row is prefetched
// (global_prefetch_b8; 32 lanes x 16B stride cover all its cachelines) so the
// data-dependent gather latency overlaps the current edge's atomics.
// ---------------------------------------------------------------------------
__global__ __launch_bounds__(256)
void k_scatter(const int* __restrict__ row, const int* __restrict__ col,
               const float* __restrict__ dinv,
               const float* __restrict__ Hs, float* __restrict__ Agg) {
    int gid  = blockIdx.x * blockDim.x + threadIdx.x;
    int wg   = __builtin_amdgcn_readfirstlane(gid >> 5);   // uniform wave id
    int lane = threadIdx.x & 31;
    int e0   = wg * EPW;
    int e1   = e0 + EPW;
    if (e1 > NE) e1 = NE;

    for (int e = e0; e < e1; ++e) {
        if (e + 1 < e1) {                 // prefetch next edge's source row
            int rn = row[e + 1];
            __builtin_prefetch(&Hs[(size_t)rn * D + lane * 4], 0, 0);
        }
        int r = row[e];
        int c = col[e];
        float w = dinv[c];
        const float4 v = ((const float4*)(Hs + (size_t)r * D))[lane];
        float* dst = Agg + (size_t)c * D + lane * 4;
        atomicAdd(dst + 0, v.x * w);
        atomicAdd(dst + 1, v.y * w);
        atomicAdd(dst + 2, v.z * w);
        atomicAdd(dst + 3, v.w * w);
    }
}

// ---------------------------------------------------------------------------
// out = relu(Agg + bias)     (between conv0 and conv1)
// ---------------------------------------------------------------------------
__global__ void k_bias_relu(const float* __restrict__ Agg,
                            const float* __restrict__ b, float* __restrict__ Out) {
    int i = blockIdx.x * blockDim.x + threadIdx.x;
    if (i < NN * D) {
        float v = Agg[i] + b[i & (D - 1)];
        Out[i] = v > 0.0f ? v : 0.0f;
    }
}

// ---------------------------------------------------------------------------
// Fused final bias+relu+global-max-pool: pooled[g,d] = max over nodes of
// relu(Agg + b). Values >= 0, so integer atomicMax on float bits is exact
// (pooled initialized to 0 == 0.0f). Skips one full 25.6MB write+read pass.
// ---------------------------------------------------------------------------
__global__ void k_pool_init(int* pooled) {
    int i = blockIdx.x * blockDim.x + threadIdx.x;
    if (i < NG * D) pooled[i] = 0;
}

__global__ void k_bias_relu_poolmax(const float* __restrict__ Agg,
                                    const float* __restrict__ b,
                                    const int* __restrict__ batch,
                                    int* __restrict__ pooled) {
    int i = blockIdx.x * blockDim.x + threadIdx.x;
    if (i < NN * D) {
        int node = i >> 7;
        int d    = i & (D - 1);
        float v  = Agg[i] + b[d];
        v = v > 0.0f ? v : 0.0f;
        int g = batch[node];
        atomicMax(&pooled[g * D + d], __float_as_int(v));
    }
}

// ---------------------------------------------------------------------------
// MLP head: relu(p@w0+b0) -> relu(t@w1+b1) -> log_softmax. One block per graph.
// ---------------------------------------------------------------------------
__global__ __launch_bounds__(128)
void k_head(const int* __restrict__ pooled,
            const float* __restrict__ w0, const float* __restrict__ b0,
            const float* __restrict__ w1, const float* __restrict__ b1,
            float* __restrict__ out) {
    __shared__ float p[D];
    __shared__ float t[D];
    __shared__ float z[NC];
    __shared__ float red[2];
    int g = blockIdx.x, tid = threadIdx.x;

    p[tid] = __int_as_float(pooled[g * D + tid]);
    __syncthreads();

    float acc = b0[tid];
    #pragma unroll 8
    for (int k = 0; k < D; ++k) acc += p[k] * w0[k * D + tid];
    t[tid] = acc > 0.0f ? acc : 0.0f;
    __syncthreads();

    if (tid < NC) {
        float a = b1[tid];
        #pragma unroll 8
        for (int k = 0; k < D; ++k) a += t[k] * w1[k * NC + tid];
        z[tid] = a > 0.0f ? a : 0.0f;
    }
    __syncthreads();

    if (tid == 0) {
        float mx = z[0];
        for (int c = 1; c < NC; ++c) mx = fmaxf(mx, z[c]);
        float s = 0.0f;
        for (int c = 0; c < NC; ++c) s += expf(z[c] - mx);
        red[0] = mx;
        red[1] = logf(s);
    }
    __syncthreads();

    if (tid < NC) out[g * NC + tid] = z[tid] - red[0] - red[1];
}

// ---------------------------------------------------------------------------
extern "C" void kernel_launch(void* const* d_in, const int* in_sizes, int n_in,
                              void* d_out, int out_size, void* d_ws, size_t ws_size,
                              hipStream_t stream) {
    (void)in_sizes; (void)n_in; (void)out_size; (void)ws_size;

    const float* x       = (const float*)d_in[0];
    const int*   edge    = (const int*)d_in[1];
    const int*   row     = edge;             // edge_index[0]
    const int*   col     = edge + NE;        // edge_index[1]
    const int*   batch   = (const int*)d_in[2];
    const float* conv0_w = (const float*)d_in[3];
    const float* conv0_b = (const float*)d_in[4];
    const float* conv1_w = (const float*)d_in[5];
    const float* conv1_b = (const float*)d_in[6];
    const float* lin0_w  = (const float*)d_in[7];
    const float* lin0_b  = (const float*)d_in[8];
    const float* lin1_w  = (const float*)d_in[9];
    const float* lin1_b  = (const float*)d_in[10];
    float* out = (float*)d_out;

    // workspace layout (floats): dinv[NN] | Hs[NN*D] | Agg[NN*D] | pooled[NG*D]
    float* dinv = (float*)d_ws;
    float* Hs   = dinv + NN;                 // 200000 B offset, 16B aligned
    float* Agg  = Hs + (size_t)NN * D;
    int*   pooled = (int*)(Agg + (size_t)NN * D);

    const int T = 256;
    const int gemm_blocks    = (NN + 31) / 32;
    const int scatter_waves  = (NE + EPW - 1) / EPW;           // 200000
    const int scatter_blocks = (scatter_waves * 32 + T - 1) / T;

    // normalization coefficients
    k_deg_init <<<(NN + T - 1) / T, T, 0, stream>>>(dinv);
    k_deg_count<<<(NE + T - 1) / T, T, 0, stream>>>(col, dinv);
    k_dinv     <<<(NN + T - 1) / T, T, 0, stream>>>(dinv);

    // ---- conv0 ----
    k_gemm_gcn <<<gemm_blocks, 128, 0, stream>>>(x, conv0_w, dinv, Hs, Agg, NN);
    k_scatter  <<<scatter_blocks, T, 0, stream>>>(row, col, dinv, Hs, Agg);
    k_bias_relu<<<(NN * D + T - 1) / T, T, 0, stream>>>(Agg, conv0_b, Hs);

    // ---- conv1 (GEMM in place on Hs) ----
    k_gemm_gcn <<<gemm_blocks, 128, 0, stream>>>(Hs, conv1_w, dinv, Hs, Agg, NN);
    k_scatter  <<<scatter_blocks, T, 0, stream>>>(row, col, dinv, Hs, Agg);

    // ---- fused bias+relu+pooling ----
    k_pool_init<<<(NG * D + T - 1) / T, T, 0, stream>>>(pooled);
    k_bias_relu_poolmax<<<(NN * D + T - 1) / T, T, 0, stream>>>(Agg, conv1_b,
                                                                batch, pooled);

    // ---- MLP head + log_softmax ----
    k_head<<<NG, 128, 0, stream>>>(pooled, lin0_w, lin0_b, lin1_w, lin1_b, out);
}
